// CatBrain_15195594293741
// MI455X (gfx1250) — compile-verified, matrix operand order checked by reading, not run on hardware
//
#include <hip/hip_runtime.h>
#include <hip/hip_bf16.h>
#include <math.h>

// ---------------------------------------------------------------------------
// CatBrain forward on gfx1250 (MI455X).
//  - recurrence: 4 blocks x 512 threads (16 waves); each block owns 16 batch
//    rows; thalamus gate/proj matmuls fused into one rolled k-loop of
//    v_wmma_f32_16x16x32_f16 with 4 independent accumulator chains
//    (rolled on purpose: full unroll spills B-fragments to scratch)
//  - A-fragments staged once per timestep as f16 in LDS (shared by all waves)
//  - d_ws: 1 MB of f16 weight fragments (prep kernel)
//  - input order assumes JAX pytree flatten (dict keys sorted at all levels)
// ---------------------------------------------------------------------------

typedef __attribute__((ext_vector_type(16))) _Float16 v16h;
typedef __attribute__((ext_vector_type(8)))  float    v8f;

#define NREG 12
#define TPB  512

// region metadata, EXECUTION (topological) order:
// 0 olfactory 1 visual 2 auditory 3 whisker 4 thalamus_relay 5 brainstem
// 6 amygdala 7 hippocampus 8 basal_ganglia 9 prefrontal 10 cerebellum 11 motor
__constant__ int   RH[NREG]    = {40,71,51,51,30,30,40,40,30,20,61,40};
__constant__ int   RSOFF[NREG] = {122,0,71,453,192,162,222,262,363,393,302,413};
__constant__ float RDT[NREG]   = {1.0f,1.8f,2.0f,2.5f,1.0f,1.0f,2.2f,1.0f,1.0f,0.8f,1.0f,1.5f};
__constant__ int   RNSEG[NREG] = {2,2,2,2,5,2,2,3,2,3,3,2};
// input segments (offset into 504-state layout; -1 = projected input buffer)
__constant__ int RSEGO[NREG][5] = {
  {-1,122,  0,  0,  0},   // olfactory: inp, self
  {-1,  0,  0,  0,  0},   // visual:    inp, self
  {-1, 71,  0,  0,  0},   // auditory
  {-1,453,  0,  0,  0},   // whisker
  {122,  0, 71,453,192},  // thalamus_relay: olf,vis,aud,whisk,self
  {192,162,  0,  0,  0},  // brainstem: thal, self
  {162,222,  0,  0,  0},  // amygdala:  brainstem, self
  {222,262,262,  0,  0},  // hippocampus: amyg, self, self (loop)
  {262,363,  0,  0,  0},  // basal_ganglia: hipp, self
  {363,393,393,  0,  0},  // prefrontal: bg, self, self (loop)
  {393,302,302,  0,  0},  // cerebellum: pf, self, self (loop)
  {302,413,  0,  0,  0},  // motor: cereb, self
};
__constant__ int RSEGL[NREG][5] = {
  {40,40, 0, 0, 0},
  {40,71, 0, 0, 0},
  {40,51, 0, 0, 0},
  {40,51, 0, 0, 0},
  {40,71,51,51,30},
  {30,30, 0, 0, 0},
  {30,40, 0, 0, 0},
  {40,40,40, 0, 0},
  {40,30, 0, 0, 0},
  {30,20,20, 0, 0},
  {20,61,61, 0, 0},
  {61,40, 0, 0, 0},
};

struct Params {
  const float* rW[NREG]; const float* rb[NREG];
  const float* rlnb[NREG]; const float* rlng[NREG]; const float* rls[NREG];
  const float* ipW; const float* ipb;
  const float* pgW; const float* pgb;
  const float* thb; const float* thlnb; const float* thlng;
  const float* tpb;
  const float* x;
};

// --------------------------------------------------------------------------
// prep: fp32 [504,512] weights -> f16 fragment-major layout for WMMA B-frags
//   frag element index = ((ntile*16 + kstep)*32 + lane)*16 + e
//   source k = kstep*32 + (e&8?16:0) + (lane&16?8:0) + (e&7), n = ntile*16+(lane&15)
// --------------------------------------------------------------------------
__global__ __launch_bounds__(512) void prep_w16(const float* __restrict__ thW,
                                                const float* __restrict__ tpW,
                                                _Float16* __restrict__ wf16) {
  int gid = blockIdx.x * blockDim.x + threadIdx.x;     // 0 .. 524287
  int mat  = gid >> 18;
  int rem  = gid & 262143;
  int e    = rem & 15;
  int lane = (rem >> 4) & 31;
  int ks   = (rem >> 9) & 15;
  int nt   = rem >> 13;
  int k = ks*32 + ((e & 8) ? 16 : 0) + ((lane & 16) ? 8 : 0) + (e & 7);
  int n = nt*16 + (lane & 15);
  const float* W = mat ? tpW : thW;
  float v = (k < 504) ? W[(size_t)k*512 + n] : 0.0f;
  wf16[gid] = (_Float16)v;
}

__device__ __forceinline__ float sigf(float x) { return 1.0f/(1.0f + expf(-x)); }

// --------------------------------------------------------------------------
// recurrence: one block per 16 batch rows, full T=512 scan
// --------------------------------------------------------------------------
__global__ __launch_bounds__(TPB) void brain_scan(Params P,
                                                  const _Float16* __restrict__ wf16,
                                                  float* __restrict__ hidden) {
  __shared__ float s[16][512];                         // 32 KB state (K padded 504->512)
  __shared__ __align__(16) char scratch[16*32*16*2];   // 16 KB: zbuf+inp (regions) / af16 (wmma)
  __shared__ float red[16][16][2];                     // per-wave LN partials
  __shared__ float stat[16][2];                        // mean / rstd per row
  float    (*zbuf)[72]     = (float    (*)[72])scratch;
  float    (*inp)[40]      = (float    (*)[40])(scratch + 16*72*4);
  _Float16 (*af16)[32][16] = (_Float16 (*)[32][16])scratch;

  const int tid  = threadIdx.x;
  const int b0   = blockIdx.x * 16;
  const int wave = tid >> 5, lane = tid & 31;
  const int m16  = lane & 15, hh = lane >> 4;
  const int nt0  = wave * 2, nt1 = nt0 + 1;
  const int cA   = nt0*16 + m16, cB = nt1*16 + m16;
  // per-lane readout constants, invariant over t
  const float biasA = P.thb[cA],   biasB = P.thb[cB];
  const float tbA   = P.tpb[cA],   tbB   = P.tpb[cB];
  const float lngA  = P.thlng[cA], lngB  = P.thlng[cB];
  const float lnbA  = P.thlnb[cA], lnbB  = P.thlnb[cB];
  // B-fragment base pointers (invariant over t); per-kstep stride = 512 halfs
  const _Float16* __restrict__ bgA = wf16 +          ((nt0*512 + lane) << 4);
  const _Float16* __restrict__ bgB = wf16 +          ((nt1*512 + lane) << 4);
  const _Float16* __restrict__ bpA = wf16 + 262144 + ((nt0*512 + lane) << 4);
  const _Float16* __restrict__ bpB = wf16 + 262144 + ((nt1*512 + lane) << 4);

  for (int i = tid; i < 16*512; i += TPB) ((float*)s)[i] = 0.0f;
  __syncthreads();

  for (int t = 0; t < 512; ++t) {
    // -- input projection: inp[16][40] = x_t @ Wip + bip
    for (int i = tid; i < 16*40; i += TPB) {
      int r = i / 40, c = i % 40;
      const float* xr = P.x + ((size_t)(b0 + r)*512 + t)*32;
      if (c == 0 && t + 1 < 512) __builtin_prefetch(xr + 32, 0, 0);  // next step's row
      float a = P.ipb[c];
      for (int k = 0; k < 32; ++k) a += xr[k] * P.ipW[k*40 + c];
      inp[r][c] = a;
    }
    __syncthreads();

    // -- region cells in topological order
    for (int reg = 0; reg < NREG; ++reg) {
      if (reg == 4 && t > 0) {
        // predatory gate on visual cortex (uses previous-step amygdala)
        for (int i = tid; i < 16*71; i += TPB) {
          int r = i / 71, c = i % 71;
          float a = P.pgb[c];
          for (int k = 0; k < 40; ++k) a += s[r][222 + k] * P.pgW[k*71 + c];
          zbuf[r][c] = 1.0f + 0.3f * sigf(a);
        }
        __syncthreads();
        for (int i = tid; i < 16*71; i += TPB) { int r = i/71, c = i%71; s[r][c] *= zbuf[r][c]; }
        __syncthreads();
      }
      const int h = RH[reg], soff = RSOFF[reg], ns = RNSEG[reg];
      const float* W  = P.rW[reg];
      const float* bb = P.rb[reg];
      // z = concat(segments) @ W + b
      for (int i = tid; i < 16*h; i += TPB) {
        int r = i / h, j = i % h;
        float a = bb[j];
        int iw = 0;
        for (int sg = 0; sg < ns; ++sg) {
          int off = RSEGO[reg][sg], len = RSEGL[reg][sg];
          const float* src = (off < 0) ? &inp[r][0] : &s[r][off];
          for (int k = 0; k < len; ++k) a += src[k] * W[(size_t)(iw + k)*h + j];
          iw += len;
        }
        zbuf[r][j] = a;
      }
      __syncthreads();
      // liquid update h_new = h + exp(ls)*dt*(tanh(z)-h)
      for (int i = tid; i < 16*h; i += TPB) {
        int r = i / h, j = i % h;
        float hv = s[r][soff + j];
        zbuf[r][j] = hv + expf(P.rls[reg][j]) * RDT[reg] * (tanhf(zbuf[r][j]) - hv);
      }
      __syncthreads();
      // LN stats per row
      if (tid < 16) {
        float m = 0.0f;
        for (int j = 0; j < h; ++j) m += zbuf[tid][j];
        m /= (float)h;
        float v = 0.0f;
        for (int j = 0; j < h; ++j) { float d = zbuf[tid][j] - m; v += d*d; }
        v /= (float)h;
        stat[tid][0] = m; stat[tid][1] = rsqrtf(v + 1e-5f);
      }
      __syncthreads();
      for (int i = tid; i < 16*h; i += TPB) {
        int r = i / h, j = i % h;
        s[r][soff + j] = (zbuf[r][j] - stat[r][0]) * stat[r][1] * P.rlng[reg][j] + P.rlnb[reg][j];
      }
      __syncthreads();
    }

    // -- stage all A-fragments (f16) once; shared by all 16 waves
    {
      int ks = wave;                         // 16 waves x 32 lanes = all (ks,lane)
      int sm = lane & 15;
      int kb = ks*32 + ((lane & 16) ? 8 : 0);
      v16h af;
#pragma unroll
      for (int e = 0; e < 16; ++e) af[e] = (_Float16)s[sm][kb + ((e & 8) ? 16 : 0) + (e & 7)];
      *(v16h*)&af16[ks][lane][0] = af;
    }
    __syncthreads();

    // -- thalamic readout: gate = sigmoid(LN(allr@Wg+bg)), out = gate*(allr@Wp+bp)
    //    single ROLLED k-loop, 4 independent WMMA accumulator chains
    {
      v8f accA{}, accB{}, pA{}, pB{};
#pragma clang loop unroll(disable)
      for (int ks = 0; ks < 16; ++ks) {
        int o = ks << 9;                               // 512 halfs per k-step
        v16h af  = *(const v16h*)&af16[ks][lane][0];
        v16h bg0 = *(const v16h*)(bgA + o);
        v16h bg1 = *(const v16h*)(bgB + o);
        v16h bp0 = *(const v16h*)(bpA + o);
        v16h bp1 = *(const v16h*)(bpB + o);
        accA = __builtin_amdgcn_wmma_f32_16x16x32_f16(false, af, false, bg0, (short)0, accA, false, false);
        accB = __builtin_amdgcn_wmma_f32_16x16x32_f16(false, af, false, bg1, (short)0, accB, false, false);
        pA   = __builtin_amdgcn_wmma_f32_16x16x32_f16(false, af, false, bp0, (short)0, pA,   false, false);
        pB   = __builtin_amdgcn_wmma_f32_16x16x32_f16(false, af, false, bp1, (short)0, pB,   false, false);
      }
      // gate bias + LN partial sums per row
      float ps[8], pq[8];
#pragma unroll
      for (int v = 0; v < 8; ++v) {
        float a = accA[v] + biasA, b = accB[v] + biasB;
        accA[v] = a; accB[v] = b;
        ps[v] = a + b; pq[v] = a*a + b*b;
      }
#pragma unroll
      for (int v = 0; v < 8; ++v)
        for (int mk = 1; mk < 16; mk <<= 1) {
          ps[v] += __shfl_xor(ps[v], mk, 32);
          pq[v] += __shfl_xor(pq[v], mk, 32);
        }
      if (m16 == 0) {
#pragma unroll
        for (int v = 0; v < 8; ++v) { red[wave][v + 8*hh][0] = ps[v]; red[wave][v + 8*hh][1] = pq[v]; }
      }
      __syncthreads();
      if (tid < 16) {
        float sm = 0.0f, sq = 0.0f;
        for (int w = 0; w < 16; ++w) { sm += red[w][tid][0]; sq += red[w][tid][1]; }
        float mn  = sm * (1.0f/512.0f);
        float var = sq * (1.0f/512.0f) - mn*mn;
        stat[tid][0] = mn; stat[tid][1] = rsqrtf(var + 1e-5f);
      }
      __syncthreads();
      // sigmoid(LN(gate)) * (proj + bias) -> hidden
#pragma unroll
      for (int v = 0; v < 8; ++v) {
        int row = v + 8*hh;
        float mn = stat[row][0], rs2 = stat[row][1];
        float gA = sigf((accA[v] - mn)*rs2*lngA + lnbA);
        float gB = sigf((accB[v] - mn)*rs2*lngB + lnbB);
        size_t base = ((size_t)(b0 + row)*512 + t)*512;
        hidden[base + cA] = (pA[v] + tbA) * gA;
        hidden[base + cB] = (pB[v] + tbB) * gB;
      }
      __syncthreads();   // protect af16/scratch & stat before next timestep
    }
  }
}

// --------------------------------------------------------------------------
// heads over every (b,t): anomaly always, next-step prediction + surprise t<511
// --------------------------------------------------------------------------
__global__ __launch_bounds__(128) void heads_kernel(const float* __restrict__ hidden,
    float* __restrict__ pred, float* __restrict__ anom, float* __restrict__ surp,
    const float* __restrict__ x,
    const float* __restrict__ aW, const float* __restrict__ ab,
    const float* __restrict__ alnb, const float* __restrict__ alng,
    const float* __restrict__ nW, const float* __restrict__ nb,
    const float* __restrict__ nlnb, const float* __restrict__ nlng) {
  const int bt = blockIdx.x, b = bt >> 9, t = bt & 511;
  const int tid = threadIdx.x;
  __shared__ float hrow[512];
  __shared__ float r1[128], r2[128];
  __shared__ float mv[2];
  const float* hp = hidden + (size_t)bt * 512;
  for (int i = tid; i < 512; i += 128) hrow[i] = hp[i];
  __syncthreads();
  float s0 = 0.0f, s1 = 0.0f;
  for (int i = tid; i < 512; i += 128) { float v = hrow[i]; s0 += v; s1 += v*v; }
  r1[tid] = s0; r2[tid] = s1; __syncthreads();
  for (int st = 64; st > 0; st >>= 1) {
    if (tid < st) { r1[tid] += r1[tid + st]; r2[tid] += r2[tid + st]; }
    __syncthreads();
  }
  if (tid == 0) {
    float m = r1[0] * (1.0f/512.0f);
    float var = r2[0] * (1.0f/512.0f) - m*m;
    mv[0] = m; mv[1] = rsqrtf(var + 1e-5f);
  }
  __syncthreads();
  const float m = mv[0], rs = mv[1];
  // anomaly head
  float pa = 0.0f;
  for (int i = tid; i < 512; i += 128) pa += ((hrow[i]-m)*rs*alng[i] + alnb[i]) * aW[i];
  r1[tid] = pa; __syncthreads();
  for (int st = 64; st > 0; st >>= 1) { if (tid < st) r1[tid] += r1[tid + st]; __syncthreads(); }
  if (tid == 0) anom[bt] = sigf(r1[0] + ab[0]);
  // next-step prediction + surprise
  if (t < 511 && tid < 32) {
    float a = nb[tid];
    for (int c = 0; c < 512; ++c) a += ((hrow[c]-m)*rs*nlng[c] + nlnb[c]) * nW[c*32 + tid];
    pred[((size_t)b*511 + t)*32 + tid] = a;
    float d = a - x[((size_t)b*512 + (t + 1))*32 + tid];
    float sq = d*d;
    for (int mk = 1; mk < 32; mk <<= 1) sq += __shfl_xor(sq, mk, 32);
    if (tid == 0) surp[(size_t)b*511 + t] = sq * (1.0f/32.0f);
  }
}

// --------------------------------------------------------------------------
// property head on hidden[:, -1]
// --------------------------------------------------------------------------
__global__ __launch_bounds__(256) void prop_kernel(const float* __restrict__ hidden,
    float* __restrict__ props,
    const float* __restrict__ plnb, const float* __restrict__ plng,
    const float* __restrict__ l1W, const float* __restrict__ l1b,
    const float* __restrict__ l2W, const float* __restrict__ l2b) {
  const int b = blockIdx.x, tid = threadIdx.x;
  __shared__ float hrow[512], hl[256], r1[256], r2[256], mv[2];
  const float* hp = hidden + ((size_t)b*512 + 511)*512;
  for (int i = tid; i < 512; i += 256) hrow[i] = hp[i];
  __syncthreads();
  float s0 = 0.0f, s1 = 0.0f;
  for (int i = tid; i < 512; i += 256) { float v = hrow[i]; s0 += v; s1 += v*v; }
  r1[tid] = s0; r2[tid] = s1; __syncthreads();
  for (int st = 128; st > 0; st >>= 1) {
    if (tid < st) { r1[tid] += r1[tid + st]; r2[tid] += r2[tid + st]; }
    __syncthreads();
  }
  if (tid == 0) {
    float m = r1[0] * (1.0f/512.0f);
    float var = r2[0] * (1.0f/512.0f) - m*m;
    mv[0] = m; mv[1] = rsqrtf(var + 1e-5f);
  }
  __syncthreads();
  const float m = mv[0], rs = mv[1];
  float a = l1b[tid];
  for (int c = 0; c < 512; ++c) a += ((hrow[c]-m)*rs*plng[c] + plnb[c]) * l1W[(size_t)c*256 + tid];
  a = a * 0.5f * (1.0f + erff(a * 0.70710678118654752f));   // exact GELU
  hl[tid] = a; __syncthreads();
  if (tid < 8) {
    float p = l2b[tid];
    for (int i = 0; i < 256; ++i) p += hl[i] * l2W[i*8 + tid];
    props[b*8 + tid] = p;
  }
}

// --------------------------------------------------------------------------
// launcher
// --------------------------------------------------------------------------
extern "C" void kernel_launch(void* const* d_in, const int* in_sizes, int n_in,
                              void* d_out, int out_size, void* d_ws, size_t ws_size,
                              hipStream_t stream) {
  (void)in_sizes; (void)n_in; (void)out_size; (void)ws_size;
  auto F = [&](int i) { return (const float*)d_in[i]; };

  // pytree-flatten (sorted dict keys) input indices:
  // 0-3  anomaly_head {W,b,ln_b,ln_g}; 4-5 input_proj {W,b};
  // 6-9  next_step_head {W,b,ln_b,ln_g}; 10-11 predatory_gate {W,b};
  // 12-17 property_head {l1.W,l1.b,l2.W,l2.b,ln_b,ln_g};
  // 18-77 regions (alphabetical) x {W,b,ln_b,ln_g,log_step};
  // 78-79 thalamic_proj {W,b}; 80-83 thalamus {W,b,ln_b,ln_g}; 84 x
  static const int APIDX[NREG] = {7,10,1,11,9,3,0,5,2,8,4,6}; // exec order -> alpha index
  Params P;
  for (int r = 0; r < NREG; ++r) {
    int base = 18 + APIDX[r]*5;
    P.rW[r]   = F(base + 0);
    P.rb[r]   = F(base + 1);
    P.rlnb[r] = F(base + 2);
    P.rlng[r] = F(base + 3);
    P.rls[r]  = F(base + 4);
  }
  P.ipW = F(4);  P.ipb = F(5);
  P.pgW = F(10); P.pgb = F(11);
  P.thb = F(81); P.thlnb = F(82); P.thlng = F(83);
  P.tpb = F(79);
  P.x   = F(84);
  const float* thW = F(80);
  const float* tpW = F(78);

  // output layout: pred | hidden | anomaly | surprise | props (all f32)
  float* out    = (float*)d_out;
  float* pred   = out;                         // 64*511*32 = 1046528
  float* hidden = out + 1046528;               // 64*512*512 = 16777216
  float* anom   = out + 17823744;              // 64*512
  float* surp   = out + 17856512;              // 64*511
  float* props  = out + 17889216;              // 64*8

  _Float16* wf16 = (_Float16*)d_ws;            // needs 2*262144*2 B = 1 MB

  prep_w16<<<1024, 512, 0, stream>>>(thW, tpW, wf16);
  brain_scan<<<4, TPB, 0, stream>>>(P, wf16, hidden);
  heads_kernel<<<64*512, 128, 0, stream>>>(hidden, pred, anom, surp, P.x,
      F(0), F(1), F(2), F(3), F(6), F(7), F(8), F(9));
  prop_kernel<<<64, 256, 0, stream>>>(hidden, props,
      F(16), F(17), F(12), F(13), F(14), F(15));
}
